// Encoding_26147760898607
// MI455X (gfx1250) — matrix operand order repaired and checked by reading, not compiled, not run on hardware
//
#include <hip/hip_runtime.h>
#include <hip/hip_bf16.h>
#include <math.h>

typedef __attribute__((ext_vector_type(16))) __bf16 v16bf;
typedef __attribute__((ext_vector_type(8)))  float  v8f;

#define Bq 32
#define Tq 2048
#define Dq 300
#define MQ (Bq * Tq)          // 65536 rows of [B*T, D]
#define NTILES 19             // ceil(300/16)
#define MTILES (MQ / 16)      // 4096
#define KI 10                 // K padded to 320 = 10 blocks of 32
#define MPAIRS (MTILES / 2)   // 2048
#define NPAIRS 10             // ceil(19/2), last pair half-valid
#define FRAG_BYTES 1024       // 32 lanes * 16 bf16 * 2 B

__device__ __forceinline__ __bf16 f2bf(float f) { return (__bf16)f; }

__device__ __forceinline__ v8f vzero() {
  v8f v = {0.f,0.f,0.f,0.f,0.f,0.f,0.f,0.f};
  return v;
}

// Per-lane fragment base pointer for tile `t` (fragments for ki are contiguous at 1 KB stride).
__device__ __forceinline__ const __bf16* fragbase(const __bf16* __restrict__ buf, int t, int lane) {
  return buf + ((size_t)t * KI * 32 + lane) * 16;
}
__device__ __forceinline__ v16bf ldfrag_off(const __bf16* __restrict__ p, int byteOff) {
  return *(const v16bf*)((const char*)p + byteOff);
}

// ---------------------------------------------------------------------------
// Pre-pack X [65536 x 300] f32 -> bf16 A-fragments, K zero-padded to 320.
// A 16x32 layout: lane half h holds K = kbase + 8h + {0..7} and kbase + 8h + 16 + {0..7}.
// One wave per (mtile, ki) fragment. 40960 waves.
__global__ __launch_bounds__(256) void pack_x(const float* __restrict__ X,
                                              __bf16* __restrict__ Xbf) {
  const int lane = threadIdx.x & 31;
  const int w = blockIdx.x * 8 + (threadIdx.x >> 5);   // 0..40959
  const int mtile = w / KI;
  const int ki = w - mtile * KI;
  const int lm = lane & 15;
  const int half8 = (lane >> 4) * 8;
  const float* __restrict__ row = X + (size_t)(mtile * 16 + lm) * Dq;
  const int k0 = ki * 32 + half8;
  v16bf r;
  if (ki < 9) {                                        // max k = 256+8+23 = 287 < 300
    float4 a0 = *(const float4*)(row + k0);
    float4 a1 = *(const float4*)(row + k0 + 4);
    float4 b0 = *(const float4*)(row + k0 + 16);
    float4 b1 = *(const float4*)(row + k0 + 20);
    r[0]=f2bf(a0.x); r[1]=f2bf(a0.y); r[2]=f2bf(a0.z);  r[3]=f2bf(a0.w);
    r[4]=f2bf(a1.x); r[5]=f2bf(a1.y); r[6]=f2bf(a1.z);  r[7]=f2bf(a1.w);
    r[8]=f2bf(b0.x); r[9]=f2bf(b0.y); r[10]=f2bf(b0.z); r[11]=f2bf(b0.w);
    r[12]=f2bf(b1.x);r[13]=f2bf(b1.y);r[14]=f2bf(b1.z); r[15]=f2bf(b1.w);
  } else {                                             // masked tail, k = 288..319
#pragma unroll
    for (int i = 0; i < 8; ++i) { int k = k0 + i;      r[i]   = (k < Dq) ? f2bf(row[k]) : f2bf(0.f); }
#pragma unroll
    for (int i = 0; i < 8; ++i) { int k = k0 + 16 + i; r[8+i] = (k < Dq) ? f2bf(row[k]) : f2bf(0.f); }
  }
  *(v16bf*)(Xbf + ((size_t)w * 32 + lane) * 16) = r;
}

// ---------------------------------------------------------------------------
// Pre-pack Wz/Wo [300 x 300] f32 -> bf16 B-fragments (y = x W^T, so B col n = W row n).
// B 32x16 layout: lane = col (lane&15), half h holds K = kbase + 8h + {0..7} and kbase + 8h + 16 + {0..7}.
// One wave per (type, ntile, ki). 380 waves.
__global__ __launch_bounds__(256) void pack_w(const float* __restrict__ Wz,
                                              const float* __restrict__ Wo,
                                              __bf16* __restrict__ Wzbf,
                                              __bf16* __restrict__ Wobf) {
  const int lane = threadIdx.x & 31;
  const int w = blockIdx.x * 8 + (threadIdx.x >> 5);
  if (w >= 2 * NTILES * KI) return;                    // 380 fragments
  const int type = w & 1;
  const int idx = w >> 1;                              // 0..189
  const int ntile = idx / KI;
  const int ki = idx - ntile * KI;
  const int lm = lane & 15;
  const int half16 = (lane >> 4) * 8;
  const int n = ntile * 16 + lm;
  const bool nv = (n < Dq);
  const float* __restrict__ row = (type ? Wo : Wz) + (size_t)(nv ? n : 0) * Dq;
  const int k0 = ki * 32 + half16;
  v16bf r;
#pragma unroll
  for (int i = 0; i < 16; ++i) {
    int k = k0 + i + (i >= 8 ? 8 : 0);
    r[i] = (nv && k < Dq) ? f2bf(row[k]) : f2bf(0.f);
  }
  __bf16* __restrict__ out = type ? Wobf : Wzbf;
  *(v16bf*)(out + ((size_t)idx * 32 + lane) * 16) = r;
}

// ---------------------------------------------------------------------------
// GEMM: one wave computes a 32x32 tile (2 M-tiles x 2 N-tiles) of BOTH
// z = tanh(X Wz^T + bz) and o = tanh(X Wo^T + bo).
// Fully unrolled K loop: 6 base pointers + immediate offsets -> no per-step
// address VALU; 12 b128 loads + 8 WMMAs per K-step, disjoint accumulators.
__device__ __forceinline__ void store_tile(const v8f& accZ, const v8f& accO,
                                           int mt, int nt, int lm, int half8,
                                           const float* __restrict__ bz,
                                           const float* __restrict__ bo,
                                           float* __restrict__ Z, float* __restrict__ O) {
  const int n = nt * 16 + lm;
  if (n < Dq) {
    const float bZ = bz[n];
    const float bO = bo[n];
#pragma unroll
    for (int r = 0; r < 8; ++r) {
      size_t idx = (size_t)(mt * 16 + r + half8) * Dq + n;
      Z[idx] = tanhf(accZ[r] + bZ);
      O[idx] = tanhf(accO[r] + bO);
    }
  }
}

__global__ __launch_bounds__(256) void gemm_zo_tanh(
    const __bf16* __restrict__ Xbf, const __bf16* __restrict__ Wzbf,
    const __bf16* __restrict__ Wobf,
    const float* __restrict__ bz, const float* __restrict__ bo,
    float* __restrict__ Z, float* __restrict__ O)
{
  const int lane = threadIdx.x & 31;
  const int tile = blockIdx.x * 8 + (threadIdx.x >> 5);  // 0..20479
  const int mpair = tile / NPAIRS;
  const int npair = tile - mpair * NPAIRS;               // fastest -> A reuse in L2/WGP$
  const int m0 = mpair * 2, m1 = m0 + 1;
  const int n0 = npair * 2, n1 = n0 + 1;
  const int n1c = (n1 < NTILES) ? n1 : (NTILES - 1);     // clamp B load for padded tile
  const int lm = lane & 15;
  const int half8 = (lane >> 4) * 8;

  // Per-lane base pointers; ki strides by 1024 B (immediate offsets after unroll).
  const __bf16* pa0 = fragbase(Xbf,  m0,  lane);
  const __bf16* pa1 = fragbase(Xbf,  m1,  lane);
  const __bf16* pp0 = fragbase(Wzbf, n0,  lane);
  const __bf16* pp1 = fragbase(Wzbf, n1c, lane);
  const __bf16* pq0 = fragbase(Wobf, n0,  lane);
  const __bf16* pq1 = fragbase(Wobf, n1c, lane);

  v8f z00 = vzero(), z01 = vzero(), z10 = vzero(), z11 = vzero();
  v8f o00 = vzero(), o01 = vzero(), o10 = vzero(), o11 = vzero();

#pragma unroll
  for (int ki = 0; ki < KI; ++ki) {
    const int off = ki * FRAG_BYTES;
    v16bf a0 = ldfrag_off(pa0, off);
    v16bf a1 = ldfrag_off(pa1, off);
    v16bf p0 = ldfrag_off(pp0, off);
    v16bf p1 = ldfrag_off(pp1, off);
    v16bf q0 = ldfrag_off(pq0, off);
    v16bf q1 = ldfrag_off(pq1, off);
    z00 = __builtin_amdgcn_wmma_f32_16x16x32_bf16(false, a0, false, p0, (short)0, z00, false, false);
    z10 = __builtin_amdgcn_wmma_f32_16x16x32_bf16(false, a1, false, p0, (short)0, z10, false, false);
    z01 = __builtin_amdgcn_wmma_f32_16x16x32_bf16(false, a0, false, p1, (short)0, z01, false, false);
    z11 = __builtin_amdgcn_wmma_f32_16x16x32_bf16(false, a1, false, p1, (short)0, z11, false, false);
    o00 = __builtin_amdgcn_wmma_f32_16x16x32_bf16(false, a0, false, q0, (short)0, o00, false, false);
    o10 = __builtin_amdgcn_wmma_f32_16x16x32_bf16(false, a1, false, q0, (short)0, o10, false, false);
    o01 = __builtin_amdgcn_wmma_f32_16x16x32_bf16(false, a0, false, q1, (short)0, o01, false, false);
    o11 = __builtin_amdgcn_wmma_f32_16x16x32_bf16(false, a1, false, q1, (short)0, o11, false, false);
  }

  store_tile(z00, o00, m0, n0, lm, half8, bz, bo, Z, O);
  store_tile(z10, o10, m1, n0, lm, half8, bz, bo, Z, O);
  store_tile(z01, o01, m0, n1, lm, half8, bz, bo, Z, O);
  store_tile(z11, o11, m1, n1, lm, half8, bz, bo, Z, O);
}

// ---------------------------------------------------------------------------
// Chunked affine scan of c_t = g*c + (1-g)*z, fused y = o*c.
// Block = 512 threads (16 waves) handles (batch b, 32 d-channels);
// wave w reduces time-chunk [128w,128w+128) to affine (A,B), LDS prefix, replay.
__global__ __launch_bounds__(512) void mru_scan_mul(
    const float* __restrict__ G, const float* __restrict__ Z,
    const float* __restrict__ O, float* __restrict__ Y)
{
  __shared__ float sA[16][32];
  __shared__ float sB[16][32];
  __shared__ float sC[16][32];

  const int lane = threadIdx.x & 31;
  const int wv   = threadIdx.x >> 5;
  const int b    = blockIdx.y;
  const int d    = blockIdx.x * 32 + lane;
  const bool dv  = (d < Dq);
  const int t0   = wv * 128;
  const size_t base = ((size_t)b * Tq) * Dq + (size_t)d + (size_t)t0 * Dq;

  float A = 1.f, Bv = 0.f;
  if (dv) {
    size_t idx = base;
    for (int t = 0; t < 128; ++t, idx += Dq) {
      float g = G[idx];
      float z = Z[idx];
      A  = g * A;
      Bv = g * Bv + (1.f - g) * z;
    }
  }
  sA[wv][lane] = A;
  sB[wv][lane] = Bv;
  __syncthreads();

  if (wv == 0) {
    float c = 0.f;
#pragma unroll
    for (int w = 0; w < 16; ++w) {
      sC[w][lane] = c;
      c = sA[w][lane] * c + sB[w][lane];
    }
  }
  __syncthreads();

  float c = sC[wv][lane];
  if (dv) {
    size_t idx = base;
    for (int t = 0; t < 128; ++t, idx += Dq) {
      float g = G[idx];
      float z = Z[idx];
      c = g * c + (1.f - g) * z;
      Y[idx] = O[idx] * c;
    }
  }
}

// ---------------------------------------------------------------------------
extern "C" void kernel_launch(void* const* d_in, const int* in_sizes, int n_in,
                              void* d_out, int out_size, void* d_ws, size_t ws_size,
                              hipStream_t stream) {
  (void)in_sizes; (void)n_in; (void)out_size; (void)ws_size;
  const float* G  = (const float*)d_in[0];   // gate_encoding   [B,T,D]
  const float* X  = (const float*)d_in[1];   // inputs_encoding [B,T,D]
  const float* Wz = (const float*)d_in[2];   // [D,D]
  const float* bz = (const float*)d_in[3];   // [D]
  const float* Wo = (const float*)d_in[4];   // [D,D]
  const float* bo = (const float*)d_in[5];   // [D]
  float* Y = (float*)d_out;                  // [B,T,D]

  char* ws = (char*)d_ws;
  float*  Zbuf = (float*)(ws);                                  // 78.6 MB
  float*  Obuf = (float*)(ws + (size_t)MQ * Dq * 4);            // 78.6 MB
  __bf16* Xbf  = (__bf16*)(ws + (size_t)MQ * Dq * 8);           // 41.9 MB
  __bf16* Wzbf = (__bf16*)(ws + (size_t)MQ * Dq * 8 + (size_t)MTILES * KI * FRAG_BYTES);
  __bf16* Wobf = (__bf16*)(ws + (size_t)MQ * Dq * 8 + (size_t)MTILES * KI * FRAG_BYTES + (size_t)NTILES * KI * FRAG_BYTES);

  // Pack weights (380 waves) and activations (40960 waves) into bf16 fragments.
  pack_w<<<dim3(48), 256, 0, stream>>>(Wz, Wo, Wzbf, Wobf);
  pack_x<<<dim3(MTILES * KI / 8), 256, 0, stream>>>(X, Xbf);

  // 2048 M-pairs * 10 N-pairs = 20480 wave-tiles; 8 waves per 256-thread block.
  gemm_zo_tanh<<<dim3(MPAIRS * NPAIRS / 8), 256, 0, stream>>>(Xbf, Wzbf, Wobf, bz, bo, Zbuf, Obuf);

  // (d-tiles, batches) = (10, 32) blocks of 512 threads.
  mru_scan_mul<<<dim3(10, Bq), 512, 0, stream>>>(G, Zbuf, Obuf, Y);
}
